// Attention_17635135717414
// MI455X (gfx1250) — matrix operand order repaired and checked by reading, not compiled, not run on hardware
//
#include <hip/hip_runtime.h>
#include <hip/hip_bf16.h>

// ---------------------------------------------------------------------------
// Causal MHA forward (QKV proj + RoPE + flash attention + out proj) for
// MI455X / gfx1250.  All matmuls run on v_wmma_f32_16x16x32_bf16.
// Shapes: N=8, T=2048, D=384, H=6, hd=64.
//
// Operand layouts (CDNA5 ISA 7.12.2):
//  A (16x32 bf16): row = lane%16; lanes 0-15 hold K={0..7,16..23},
//                  lanes 16-31 hold K={8..15,24..31} (pairs packed per VGPR).
//  B (32x16 bf16): col = lane%16; K order as for A; this code loads A and B
//                  both contiguous: lanes 0-15 K=0..15, 16-31 K=16..31.
//  C/D (16x16 f32): VGPR r holds row M = r + 8*(lane/16), col N = lane%16.
// ---------------------------------------------------------------------------

typedef __attribute__((ext_vector_type(16))) __bf16 v16bf;
typedef __attribute__((ext_vector_type(8)))  __bf16 v8bf;
typedef __attribute__((ext_vector_type(8)))  float  v8f;

#define NB 8
#define TT 2048
#define DD 384
#define HH 6
#define HD 64
#define E3D (3 * DD)          // 1152

// A-operand fragment: 16 contiguous K values starting at kk + half*16.
static __device__ inline v16bf load_fragA(const __bf16* rowptr, int kk, int half) {
  v8bf a = *reinterpret_cast<const v8bf*>(rowptr + kk + half * 16);
  v8bf b = *reinterpret_cast<const v8bf*>(rowptr + kk + half * 16 + 8);
  v16bf r;
#pragma unroll
  for (int i = 0; i < 8; ++i) { r[i] = a[i]; r[i + 8] = b[i]; }
  return r;
}

// B-operand fragment: 16 contiguous K values starting at kk + half*16.
static __device__ inline v16bf load_fragB(const __bf16* rowptr, int kk, int half) {
  v8bf a = *reinterpret_cast<const v8bf*>(rowptr + kk + half * 16);
  v8bf b = *reinterpret_cast<const v8bf*>(rowptr + kk + half * 16 + 8);
  v16bf r;
#pragma unroll
  for (int i = 0; i < 8; ++i) { r[i] = a[i]; r[i + 8] = b[i]; }
  return r;
}

// ---------------------------------------------------------------------------
// Kernel 1: f32 -> bf16 conversion
// ---------------------------------------------------------------------------
__global__ void k_f32_to_bf16(const float* __restrict__ in, __bf16* __restrict__ out, int n) {
  int i = blockIdx.x * blockDim.x + threadIdx.x;
  if (i < n) out[i] = (__bf16)in[i];
}

// ---------------------------------------------------------------------------
// Kernel 2: QKV projection.  qkv[g,e] = sum_d x[g,d] * Wqkv[e,d]
// One wave per 16x16 output tile; epilogue scatters into Q/K/V [N,H,T,hd].
// ---------------------------------------------------------------------------
__global__ __launch_bounds__(256) void k_qkv_gemm(const __bf16* __restrict__ xb,
                                                  const __bf16* __restrict__ wb,
                                                  __bf16* __restrict__ Qo,
                                                  __bf16* __restrict__ Ko,
                                                  __bf16* __restrict__ Vo) {
  const int lane = threadIdx.x & 31;
  const int half = lane >> 4, l16 = lane & 15;
  const int wid = blockIdx.x * 8 + (threadIdx.x >> 5);
  const int ntiles = E3D / 16;                 // 72
  const int mt = wid / ntiles, nt = wid % ntiles;

  const __bf16* arow = xb + (size_t)(mt * 16 + l16) * DD;
  const __bf16* brow = wb + (size_t)(nt * 16 + l16) * DD;

  v8f acc = {0.f, 0.f, 0.f, 0.f, 0.f, 0.f, 0.f, 0.f};
#pragma unroll 4
  for (int kk = 0; kk < DD; kk += 32) {
    v16bf a = load_fragA(arow, kk, half);
    v16bf b = load_fragB(brow, kk, half);
    acc = __builtin_amdgcn_wmma_f32_16x16x32_bf16(false, a, false, b, (short)0, acc, false, false);
  }

  const int e = nt * 16 + l16;
  const int which = e / DD;
  const int w2 = e - which * DD;
  const int h = w2 >> 6, d = w2 & 63;
  __bf16* dst = (which == 0) ? Qo : (which == 1 ? Ko : Vo);
#pragma unroll
  for (int r = 0; r < 8; ++r) {
    const int g = mt * 16 + r + 8 * half;       // global token index
    const int n = g >> 11, t = g & (TT - 1);
    dst[((size_t)((n * HH + h) * TT + t)) * HD + d] = (__bf16)acc[r];
  }
}

// ---------------------------------------------------------------------------
// Kernel 3: RoPE, in-place on Q and K.  One thread per (n,h,t,i<32) pair.
// ---------------------------------------------------------------------------
__global__ void k_rope(__bf16* __restrict__ Q, __bf16* __restrict__ K) {
  const int tid = blockIdx.x * blockDim.x + threadIdx.x;
  const int i = tid & 31;
  const int t = (tid >> 5) & (TT - 1);
  const int nh = tid >> 16;                     // / (32*2048)
  if (nh >= NB * HH) return;

  const float ang = (float)t * __powf(10000.0f, -(float)i * (1.0f / 32.0f));
  float s, c;
  __sincosf(ang, &s, &c);

  const size_t base = ((size_t)nh * TT + t) * HD;
  float q1 = (float)Q[base + i], q2 = (float)Q[base + i + 32];
  Q[base + i]      = (__bf16)(q1 * c - q2 * s);
  Q[base + i + 32] = (__bf16)(q2 * c + q1 * s);
  float k1 = (float)K[base + i], k2 = (float)K[base + i + 32];
  K[base + i]      = (__bf16)(k1 * c - k2 * s);
  K[base + i + 32] = (__bf16)(k2 * c + k1 * s);
}

// ---------------------------------------------------------------------------
// Kernel 4: causal flash attention.  One wave per (n, h, 16-query tile).
// Online softmax over 32-key steps.  LDS roundtrip converts P from the
// C-matrix layout to an A-matrix fragment (fenced with s_wait_dscnt).
// ---------------------------------------------------------------------------
__global__ __launch_bounds__(128) void k_attn(const __bf16* __restrict__ Q,
                                              const __bf16* __restrict__ K,
                                              const __bf16* __restrict__ V,
                                              __bf16* __restrict__ ctx) {
  __shared__ __bf16 plds[4][16][32];            // 4 KB: one 16x32 P tile per wave
  const int lane = threadIdx.x & 31;
  const int wv = threadIdx.x >> 5;
  const int half = lane >> 4, l16 = lane & 15;

  const int QT = TT / 16;                       // 128
  const int tile = blockIdx.x * 4 + wv;
  const int n = tile / (HH * QT);
  const int rem = tile - n * (HH * QT);
  const int h = rem / QT, qt = rem % QT;
  const int q0 = qt * 16, qlast = q0 + 15;

  const size_t hb = ((size_t)(n * HH + h)) * TT * HD;
  const __bf16* Qh = Q + hb;
  const __bf16* Kh = K + hb;
  const __bf16* Vh = V + hb;

  // Q fragments (A operand): two 16x32 chunks covering hd=64
  const __bf16* qrow = Qh + (size_t)(q0 + l16) * HD;
  const v16bf qa0 = load_fragA(qrow, 0, half);
  const v16bf qa1 = load_fragA(qrow, 32, half);

  float rm[8], rs[8];
  v8f o[4];
#pragma unroll
  for (int r = 0; r < 8; ++r) {
    rm[r] = -1e30f;
    rs[r] = 0.0f;
#pragma unroll
    for (int dt = 0; dt < 4; ++dt) o[dt][r] = 0.0f;
  }
  const float SC = 0.125f;                      // 1/sqrt(64)

  for (int kb = 0; kb <= qlast; kb += 32) {
    // ---- S = Q * K^T for two 16-key subtiles (K rows are B operands) -----
    v8f s0 = {0.f, 0.f, 0.f, 0.f, 0.f, 0.f, 0.f, 0.f};
    v8f s1 = {0.f, 0.f, 0.f, 0.f, 0.f, 0.f, 0.f, 0.f};
    {
      const int keyr = min(kb + l16, TT - 1);
      const __bf16* krow = Kh + (size_t)keyr * HD;
      __builtin_prefetch(krow + 32 * HD, 0, 1); // next key tile -> global_prefetch_b8
      v16bf b0 = load_fragB(krow, 0, half);
      v16bf b1 = load_fragB(krow, 32, half);
      s0 = __builtin_amdgcn_wmma_f32_16x16x32_bf16(false, qa0, false, b0, (short)0, s0, false, false);
      s0 = __builtin_amdgcn_wmma_f32_16x16x32_bf16(false, qa1, false, b1, (short)0, s0, false, false);
    }
    {
      const int keyr = min(kb + 16 + l16, TT - 1);
      const __bf16* krow = Kh + (size_t)keyr * HD;
      v16bf b0 = load_fragB(krow, 0, half);
      v16bf b1 = load_fragB(krow, 32, half);
      s1 = __builtin_amdgcn_wmma_f32_16x16x32_bf16(false, qa0, false, b0, (short)0, s1, false, false);
      s1 = __builtin_amdgcn_wmma_f32_16x16x32_bf16(false, qa1, false, b1, (short)0, s1, false, false);
    }

    // ---- online softmax (C layout: row m = r + 8*half, col = l16) --------
    float nm[8];
#pragma unroll
    for (int r = 0; r < 8; ++r) {
      const int qi = q0 + r + 8 * half;
      float a0 = s0[r] * SC; if (kb + l16 > qi)      a0 = -1e30f;
      float a1 = s1[r] * SC; if (kb + 16 + l16 > qi) a1 = -1e30f;
      s0[r] = a0; s1[r] = a1;
      float mx = fmaxf(rm[r], fmaxf(a0, a1));
#pragma unroll
      for (int dsh = 1; dsh < 16; dsh <<= 1) mx = fmaxf(mx, __shfl_xor(mx, dsh, 32));
      nm[r] = mx;
    }
#pragma unroll
    for (int r = 0; r < 8; ++r) {
      const float al = __expf(rm[r] - nm[r]);
      const float p0 = __expf(s0[r] - nm[r]);
      const float p1 = __expf(s1[r] - nm[r]);
      float pr = p0 + p1;
#pragma unroll
      for (int dsh = 1; dsh < 16; dsh <<= 1) pr += __shfl_xor(pr, dsh, 32);
      rs[r] = rs[r] * al + pr;
      rm[r] = nm[r];
#pragma unroll
      for (int dt = 0; dt < 4; ++dt) o[dt][r] *= al;
      plds[wv][r + 8 * half][l16]      = (__bf16)p0;
      plds[wv][r + 8 * half][16 + l16] = (__bf16)p1;
    }
    // DS ops are in-order per wave; wait + compiler fence before re-reading.
    asm volatile("s_wait_dscnt 0" ::: "memory");

    // ---- P (A fragment from LDS) * V (B fragments, gathered) -------------
    const v16bf pa = load_fragA(&plds[wv][l16][0], 0, half);
#pragma unroll
    for (int dt = 0; dt < 4; ++dt) {
      v16bf vb;
      const int dcol = dt * 16 + l16;
#pragma unroll
      for (int i = 0; i < 16; ++i) {
        const int key = kb + half * 16 + i;     // B layout: contiguous K per half
        const int kr = min(key, TT - 1);
        vb[i] = Vh[(size_t)kr * HD + dcol];
      }
      o[dt] = __builtin_amdgcn_wmma_f32_16x16x32_bf16(false, pa, false, vb, (short)0, o[dt], false, false);
    }
  }

  // ---- finalize: ctx[g, h*64 + d] ----------------------------------------
#pragma unroll
  for (int r = 0; r < 8; ++r) {
    const float inv = 1.0f / rs[r];
    const int g = n * TT + q0 + r + 8 * half;
#pragma unroll
    for (int dt = 0; dt < 4; ++dt)
      ctx[(size_t)g * DD + h * HD + dt * 16 + l16] = (__bf16)(o[dt][r] * inv);
  }
}

// ---------------------------------------------------------------------------
// Kernel 5: output projection.  out[g,e] = sum_d ctx[g,d] * Wout[e,d]  (f32 out)
// ---------------------------------------------------------------------------
__global__ __launch_bounds__(256) void k_out_gemm(const __bf16* __restrict__ cb,
                                                  const __bf16* __restrict__ wb,
                                                  float* __restrict__ out) {
  const int lane = threadIdx.x & 31;
  const int half = lane >> 4, l16 = lane & 15;
  const int wid = blockIdx.x * 8 + (threadIdx.x >> 5);
  const int ntiles = DD / 16;                   // 24
  const int mt = wid / ntiles, nt = wid % ntiles;

  const __bf16* arow = cb + (size_t)(mt * 16 + l16) * DD;
  const __bf16* brow = wb + (size_t)(nt * 16 + l16) * DD;

  v8f acc = {0.f, 0.f, 0.f, 0.f, 0.f, 0.f, 0.f, 0.f};
#pragma unroll 4
  for (int kk = 0; kk < DD; kk += 32) {
    v16bf a = load_fragA(arow, kk, half);
    v16bf b = load_fragB(brow, kk, half);
    acc = __builtin_amdgcn_wmma_f32_16x16x32_bf16(false, a, false, b, (short)0, acc, false, false);
  }

  const int e = nt * 16 + l16;
#pragma unroll
  for (int r = 0; r < 8; ++r) {
    const int g = mt * 16 + r + 8 * half;
    out[(size_t)g * DD + e] = acc[r];
  }
}

// ---------------------------------------------------------------------------
// Host-side launch
// ---------------------------------------------------------------------------
extern "C" void kernel_launch(void* const* d_in, const int* in_sizes, int n_in,
                              void* d_out, int out_size, void* d_ws, size_t ws_size,
                              hipStream_t stream) {
  (void)in_sizes; (void)n_in; (void)out_size; (void)ws_size;
  const float* x    = (const float*)d_in[0];   // [8,2048,384]
  const float* Wqkv = (const float*)d_in[1];   // [1152,384]
  const float* Wout = (const float*)d_in[2];   // [384,384]
  float* out = (float*)d_out;                  // [8,2048,384]

  const size_t XB   = (size_t)NB * TT * DD;        // 6,291,456
  const size_t WQ   = (size_t)E3D * DD;            //   442,368
  const size_t WO   = (size_t)DD * DD;             //   147,456
  const size_t HB   = (size_t)NB * HH * TT * HD;   // 6,291,456

  char* ws = (char*)d_ws;
  size_t off = 0;
  __bf16* xb    = (__bf16*)(ws + off); off += XB * 2;
  __bf16* wqkvb = (__bf16*)(ws + off); off += WQ * 2;
  __bf16* woutb = (__bf16*)(ws + off); off += WO * 2;
  __bf16* Qb    = (__bf16*)(ws + off); off += HB * 2;
  __bf16* Kb    = (__bf16*)(ws + off); off += HB * 2;
  __bf16* Vb    = (__bf16*)(ws + off); off += HB * 2;
  __bf16* ctxb  = (__bf16*)(ws + off); off += XB * 2;

  // 1) convert f32 -> bf16
  k_f32_to_bf16<<<(int)((XB + 255) / 256), 256, 0, stream>>>(x, xb, (int)XB);
  k_f32_to_bf16<<<(int)((WQ + 255) / 256), 256, 0, stream>>>(Wqkv, wqkvb, (int)WQ);
  k_f32_to_bf16<<<(int)((WO + 255) / 256), 256, 0, stream>>>(Wout, woutb, (int)WO);

  // 2) QKV projection: 1024 * 72 wave-tiles, 8 waves/block
  k_qkv_gemm<<<(NB * TT / 16) * (E3D / 16) / 8, 256, 0, stream>>>(xb, wqkvb, Qb, Kb, Vb);

  // 3) RoPE: NB*HH*TT*32 threads
  k_rope<<<NB * HH * TT * 32 / 256, 256, 0, stream>>>(Qb, Kb);

  // 4) attention: NB*HH*(TT/16) wave-tiles, 4 waves/block
  k_attn<<<NB * HH * (TT / 16) / 4, 128, 0, stream>>>(Qb, Kb, Vb, ctxb);

  // 5) output projection: 1024 * 24 wave-tiles, 8 waves/block
  k_out_gemm<<<(NB * TT / 16) * (DD / 16) / 8, 256, 0, stream>>>(ctxb, woutb, out);
}
